// SelfAttention_25864293057288
// MI455X (gfx1250) — compile-verified, matrix operand order checked by reading, not run on hardware
//
#include <hip/hip_runtime.h>
#include <hip/hip_bf16.h>

// ---------------------------------------------------------------------------
// SelfAttention, B=16 C=64 H=1024 W=16  (D = 128, seq N = 1024)
//   Q [b][p][i]   = dot(Wq[i/16], X[b,:, (i%16)*1024 + p]) + bq[i/16]
//   Kt[b][p'][i]  = dot(Wk[i/16], X[b,:, (i%16)*1024 + p']) + bk[i/16]
//   Vt[b][s][r]   = dot(Wv[r/16], X[b,:, (r%16)*1024 + s]) + bv[r/16]
//   S = softmax(Q Kt^T);  out = x + S Vt^T  (flat identity reshape)
// ---------------------------------------------------------------------------

typedef __attribute__((ext_vector_type(16))) _Float16 v16h;
typedef __attribute__((ext_vector_type(8)))  _Float16 v8h;
typedef __attribute__((ext_vector_type(8)))  float    v8f;
typedef int v4i_ __attribute__((vector_size(16)));

#define NB 16
#define NC 64
#define ND 128
#define NN 1024

// ---- optional gfx1250 async global->LDS path (guarded; sync fallback) -----
#if defined(__has_builtin)
#if __has_builtin(__builtin_amdgcn_global_load_async_to_lds_b128) && \
    __has_builtin(__builtin_amdgcn_s_wait_asynccnt)
#define USE_ASYNC_LDS 1
#endif
#endif

__device__ inline void cp16(void* lds_dst, const void* gsrc) {
#ifdef USE_ASYNC_LDS
  void* g = const_cast<void*>(gsrc);
  __builtin_amdgcn_global_load_async_to_lds_b128(
      (__attribute__((address_space(1))) v4i_*)g,
      (__attribute__((address_space(3))) v4i_*)lds_dst, 0, 0);
#else
  *(int4*)lds_dst = *(const int4*)gsrc;
#endif
}
__device__ inline void cp_wait() {
#ifdef USE_ASYNC_LDS
  __builtin_amdgcn_s_wait_asynccnt(0);
#endif
}

__device__ inline v8f vzero8() {
  v8f z;
#pragma unroll
  for (int i = 0; i < 8; ++i) z[i] = 0.0f;
  return z;
}

// Fragment for v_wmma_f32_16x16x32_f16 from a row-major [row][K] array:
// lane l holds row (l&15), K-halves {kb..kb+7} U {kb+16..kb+23}, kb=(l>>4)*8.
// p must be 16B aligned (row strides here are multiples of 16B).
__device__ inline v16h load_frag(const _Float16* __restrict__ p) {
  v8h lo = *(const v8h*)(p);
  v8h hi = *(const v8h*)(p + 16);
  v16h r;
#pragma unroll
  for (int i = 0; i < 8; ++i) { r[i] = lo[i]; r[i + 8] = hi[i]; }
  return r;
}

__device__ inline unsigned pack_h2(float a, float b) {
  union { _Float16 h[2]; unsigned u; } u_;
  u_.h[0] = (_Float16)a; u_.h[1] = (_Float16)b;
  return u_.u;
}

// ---------------------------------------------------------------------------
// Kernel 1: QKV projection via WMMA. One WG = (b, 8-wide s-slab), pixels
// t = jh*8+sl (128 pixels = 8 M-tiles), 4 waves x 2 M-tiles each.
// B-operands: [Wq;Wk] packed as one 16-col tile, Wv as 4 tiles. K = 64 ch.
// Outputs scattered into LDS tiles -> coalesced f16 row stores.
// ---------------------------------------------------------------------------
__global__ __launch_bounds__(128) void proj_kernel(
    const float* __restrict__ x,
    const float* __restrict__ Wq, const float* __restrict__ bq,
    const float* __restrict__ Wk, const float* __restrict__ bk,
    const float* __restrict__ Wv, const float* __restrict__ bv,
    _Float16* __restrict__ Qh, _Float16* __restrict__ Kt,
    _Float16* __restrict__ Vt) {
  __shared__ __align__(16) _Float16 Xs[128][72];   // 18 KB  (pixels x ch)
  __shared__ __align__(16) _Float16 Wqk[16][72];   // rows: o<8 Wq, o>=8 Wk
  __shared__ __align__(16) _Float16 Wvs[64][72];
  __shared__ float bqk[16];
  __shared__ float bvs[64];
  __shared__ _Float16 Qt_t[8][136];
  __shared__ _Float16 Kt_t[8][136];
  __shared__ _Float16 Vt_t[8][1032];

  const int b  = blockIdx.x >> 7;
  const int s0 = (blockIdx.x & 127) << 3;
  const int t  = threadIdx.x;
  const int jh = t >> 3, sl = t & 7;
  const int j  = jh * 1024 + s0 + sl;   // pixel index within batch

  for (int idx = t; idx < 16 * NC; idx += 128) {
    int o = idx >> 6, c = idx & 63;
    Wqk[o][c] = (_Float16)(o < 8 ? Wq[o * NC + c] : Wk[(o - 8) * NC + c]);
  }
  for (int idx = t; idx < NC * NC; idx += 128)
    Wvs[idx >> 6][idx & 63] = (_Float16)Wv[idx];
  if (t < 16) bqk[t] = (t < 8) ? bq[t] : bk[t - 8];
  if (t < 64) bvs[t] = bv[t];

  const float* xb = x + ((size_t)b << 20);
#pragma unroll
  for (int c = 0; c < NC; ++c) Xs[t][c] = (_Float16)xb[c * 16384 + j];
  __syncthreads();

  const int wave = t >> 5, lane = t & 31;
  const int m = lane & 15, hf = lane >> 4, kb = hf * 8;

#pragma unroll
  for (int mi = 0; mi < 2; ++mi) {
    const int mt = wave * 2 + mi;
    v16h a0 = load_frag(&Xs[mt * 16 + m][kb]);
    v16h a1 = load_frag(&Xs[mt * 16 + m][32 + kb]);

    v16h wq0 = load_frag(&Wqk[m][kb]);
    v16h wq1 = load_frag(&Wqk[m][32 + kb]);
    v8f accqk = vzero8();
    accqk = __builtin_amdgcn_wmma_f32_16x16x32_f16(false, a0, false, wq0,
                                                   (short)0, accqk, false, false);
    accqk = __builtin_amdgcn_wmma_f32_16x16x32_f16(false, a1, false, wq1,
                                                   (short)0, accqk, false, false);
    v8f accv[4];
#pragma unroll
    for (int nt = 0; nt < 4; ++nt) {
      v16h wv0 = load_frag(&Wvs[nt * 16 + m][kb]);
      v16h wv1 = load_frag(&Wvs[nt * 16 + m][32 + kb]);
      accv[nt] = vzero8();
      accv[nt] = __builtin_amdgcn_wmma_f32_16x16x32_f16(
          false, a0, false, wv0, (short)0, accv[nt], false, false);
      accv[nt] = __builtin_amdgcn_wmma_f32_16x16x32_f16(
          false, a1, false, wv1, (short)0, accv[nt], false, false);
    }
    // D layout: vgpr r -> pixel row (mt*16 + r + 8*hf), col = lane&15
#pragma unroll
    for (int r = 0; r < 8; ++r) {
      int tp = mt * 16 + r + 8 * hf;
      int jp = tp >> 3, sp = tp & 7;
      float vq = accqk[r] + bqk[m];
      if (m < 8) Qt_t[sp][jp + 16 * m] = (_Float16)vq;
      else       Kt_t[sp][jp + 16 * (m - 8)] = (_Float16)vq;
#pragma unroll
      for (int nt = 0; nt < 4; ++nt) {
        int cv = nt * 16 + m;
        Vt_t[sp][jp + 16 * cv] = (_Float16)(accv[nt][r] + bvs[cv]);
      }
    }
  }
  __syncthreads();

  // Coalesced stores: Q/K 8x128 halves (512 u32), V 8x1024 halves (4096 u32)
  for (int f = t; f < 8 * (ND / 2); f += 128) {
    int r = f >> 6, c2 = (f & 63) * 2;
    ((unsigned*)(Qh + (size_t)(b * NN + s0 + r) * ND))[f & 63] =
        pack_h2((float)Qt_t[r][c2], (float)Qt_t[r][c2 + 1]);
    ((unsigned*)(Kt + (size_t)(b * NN + s0 + r) * ND))[f & 63] =
        pack_h2((float)Kt_t[r][c2], (float)Kt_t[r][c2 + 1]);
  }
  for (int f = t; f < 8 * (NN / 2); f += 128) {
    int r = f >> 9, c2 = (f & 511) * 2;
    ((unsigned*)(Vt + (size_t)(b * NN + s0 + r) * NN))[f & 511] =
        pack_h2((float)Vt_t[r][c2], (float)Vt_t[r][c2 + 1]);
  }
}

// ---------------------------------------------------------------------------
// Kernel 2: S = softmax(Q Kt^T). WG = (b, 16-row block), 4 waves each owning a
// 256-col quarter. All 4 K-fragments of a tile loaded into independent regs
// before the 4 chained WMMAs (memory-level parallelism).
// ---------------------------------------------------------------------------
__global__ __launch_bounds__(128) void scores_kernel(
    const _Float16* __restrict__ Qh, const _Float16* __restrict__ Kt,
    _Float16* __restrict__ Sh) {
  __shared__ float S[16][NN];  // 64 KB

  const int b = blockIdx.x >> 6;
  const int p0 = (blockIdx.x & 63) << 4;
  const int tid = threadIdx.x;
  const int wave = tid >> 5, lane = tid & 31;
  const int m = lane & 15, hf = lane >> 4, kb = hf * 8;

  const _Float16* Qb = Qh + ((size_t)(b * NN + p0 + m)) * ND + kb;
  v16h afr[4];
#pragma unroll
  for (int ks = 0; ks < 4; ++ks) afr[ks] = load_frag(Qb + ks * 32);

  v8f acc[16];
#pragma unroll
  for (int i = 0; i < 16; ++i) acc[i] = vzero8();

  const _Float16* Ktb = Kt + ((size_t)b << 17);
  const int c0 = wave << 8;
#pragma unroll
  for (int ct = 0; ct < 16; ++ct) {
    const _Float16* Kr = Ktb + (size_t)(c0 + ct * 16 + m) * ND + kb;
    v16h bf0 = load_frag(Kr);
    v16h bf1 = load_frag(Kr + 32);
    v16h bf2 = load_frag(Kr + 64);
    v16h bf3 = load_frag(Kr + 96);
    acc[ct] = __builtin_amdgcn_wmma_f32_16x16x32_f16(false, afr[0], false, bf0,
                                                     (short)0, acc[ct], false, false);
    acc[ct] = __builtin_amdgcn_wmma_f32_16x16x32_f16(false, afr[1], false, bf1,
                                                     (short)0, acc[ct], false, false);
    acc[ct] = __builtin_amdgcn_wmma_f32_16x16x32_f16(false, afr[2], false, bf2,
                                                     (short)0, acc[ct], false, false);
    acc[ct] = __builtin_amdgcn_wmma_f32_16x16x32_f16(false, afr[3], false, bf3,
                                                     (short)0, acc[ct], false, false);
  }
#pragma unroll
  for (int ct = 0; ct < 16; ++ct)
#pragma unroll
    for (int r = 0; r < 8; ++r)
      S[r + 8 * hf][c0 + ct * 16 + m] = acc[ct][r];
  __syncthreads();

  // Row softmax: 8 lanes per row, interleaved columns q8 + 8k (bank-spread).
  const int row = tid >> 3, q8 = tid & 7;
  float mx = -3.4e38f;
  for (int k = 0; k < 128; ++k) mx = fmaxf(mx, S[row][q8 + 8 * k]);
#pragma unroll
  for (int off = 4; off; off >>= 1) mx = fmaxf(mx, __shfl_xor(mx, off, 8));
  float sum = 0.0f;
  for (int k = 0; k < 128; ++k) {
    float e = __expf(S[row][q8 + 8 * k] - mx);
    sum += e;
    S[row][q8 + 8 * k] = e;
  }
#pragma unroll
  for (int off = 4; off; off >>= 1) sum += __shfl_xor(sum, off, 8);
  const float inv = 1.0f / sum;
  for (int k = 0; k < 128; ++k) S[row][q8 + 8 * k] *= inv;
  __syncthreads();

  unsigned* dst = (unsigned*)(Sh + ((size_t)(b * NN + p0)) * NN);
  for (int f = tid; f < 16 * (NN / 2); f += 128) {
    int r2 = f >> 9, c2 = (f & 511) * 2;
    dst[(size_t)r2 * 512 + (f & 511)] = pack_h2(S[r2][c2], S[r2][c2 + 1]);
  }
}

// ---------------------------------------------------------------------------
// Kernel 3: out = x + P Vt^T. WG = (b, 64 rows, 256 cols); 4 waves = 4 row
// tiles sharing one LDS-staged V slice. Double-buffered async global->LDS
// copies (ASYNCcnt) overlap the WMMA stream; B/A fragments via ds_load_b128.
// Rows padded to 48 halves (96 B): 16B-aligned fragments, <=2-way conflicts.
// ---------------------------------------------------------------------------
__global__ __launch_bounds__(128) void attnv_kernel(
    const _Float16* __restrict__ Sh, const _Float16* __restrict__ Vt,
    const float* __restrict__ x, float* __restrict__ out) {
  __shared__ __align__(16) _Float16 Vs[2][256][48];  // 48 KB
  __shared__ __align__(16) _Float16 Ps[2][64][48];   // 12 KB

  const int b    = blockIdx.x >> 6;
  const int pblk = (blockIdx.x >> 2) & 15;
  const int cblk = blockIdx.x & 3;
  const int tid = threadIdx.x, wave = tid >> 5, lane = tid & 31;
  const int m = lane & 15, hf = lane >> 4, kb = hf * 8;
  const int p0 = pblk * 64;
  const int c0 = cblk * 256;

  const _Float16* Pb = Sh + ((size_t)(b * NN + p0)) * NN;
  const _Float16* Vb = Vt + ((size_t)b << 20) + (size_t)c0 * NN;

  v8f acc[16];
#pragma unroll
  for (int i = 0; i < 16; ++i) acc[i] = vzero8();

  auto stage = [&](int ks, int bufi) {
#pragma unroll
    for (int i = 0; i < 8; ++i) {            // V: 256 cols x 64 B
      int idx = i * 128 + tid;
      int col = idx >> 2, seg = idx & 3;
      cp16(&Vs[bufi][col][seg * 8], Vb + (size_t)col * NN + ks * 32 + seg * 8);
    }
#pragma unroll
    for (int i = 0; i < 2; ++i) {            // P: 64 rows x 64 B
      int idx = i * 128 + tid;
      int row = idx >> 2, seg = idx & 3;
      cp16(&Ps[bufi][row][seg * 8], Pb + (size_t)row * NN + ks * 32 + seg * 8);
    }
  };

  int buf = 0;
  stage(0, 0);
  cp_wait();
  __syncthreads();
  for (int ks = 0; ks < 32; ++ks) {
    if (ks < 31) stage(ks + 1, buf ^ 1);
    v16h a = load_frag(&Ps[buf][wave * 16 + m][kb]);
#pragma unroll
    for (int ct = 0; ct < 16; ++ct) {
      v16h bf = load_frag(&Vs[buf][ct * 16 + m][kb]);
      acc[ct] = __builtin_amdgcn_wmma_f32_16x16x32_f16(
          false, a, false, bf, (short)0, acc[ct], false, false);
    }
    cp_wait();
    __syncthreads();
    buf ^= 1;
  }

  const int pr0 = p0 + wave * 16;
  const float* xb = x + ((size_t)b << 20);
  float* ob = out + ((size_t)b << 20);
#pragma unroll
  for (int ct = 0; ct < 16; ++ct) {
    int s = c0 + ct * 16 + m;
#pragma unroll
    for (int r = 0; r < 8; ++r) {
      int p = pr0 + r + 8 * hf;
      size_t idx = ((size_t)p << 10) + s;
      ob[idx] = acc[ct][r] + xb[idx];
    }
  }
}

// ---------------------------------------------------------------------------
extern "C" void kernel_launch(void* const* d_in, const int* in_sizes, int n_in,
                              void* d_out, int out_size, void* d_ws,
                              size_t ws_size, hipStream_t stream) {
  const float* x  = (const float*)d_in[0];
  const float* Wq = (const float*)d_in[1];
  const float* bq = (const float*)d_in[2];
  const float* Wk = (const float*)d_in[3];
  const float* bk = (const float*)d_in[4];
  const float* Wv = (const float*)d_in[5];
  const float* bv = (const float*)d_in[6];
  float* out = (float*)d_out;

  // Workspace: Qh 4MB | Kt 4MB | Vt 32MB | Sh 32MB (72 MB f16)
  _Float16* Qh = (_Float16*)d_ws;
  _Float16* Kt = Qh + (size_t)NB * NN * ND;
  _Float16* Vt = Kt + (size_t)NB * NN * ND;
  _Float16* Sh = Vt + (size_t)NB * NN * NN;

  proj_kernel<<<NB * 128, 128, 0, stream>>>(x, Wq, bq, Wk, bk, Wv, bv,
                                            Qh, Kt, Vt);
  scores_kernel<<<NB * 64, 128, 0, stream>>>(Qh, Kt, Sh);
  attnv_kernel<<<NB * 64, 128, 0, stream>>>(Sh, Vt, x, out);
}
